// FastKANLayer_18854906429776
// MI455X (gfx1250) — compile-verified, matrix operand order checked by reading, not run on hardware
//
#include <hip/hip_runtime.h>
#include <math.h>

// Problem constants (FastKANLayer reference)
#define B_    8
#define NIN_  32
#define NOUT_ 32
#define F_    187
#define FP_   192      // F padded to multiple of 16 for WMMA tiles
#define HID_  32
#define NC_   35

typedef __attribute__((ext_vector_type(16))) __bf16 v16bf;
typedef __attribute__((ext_vector_type(8)))  float  v8f;

union FragU { uint4 q[2]; v16bf v; };

__device__ inline unsigned short f2bf(float x) {
  unsigned u = __float_as_uint(x);
  unsigned r = (u + 0x7FFFu + ((u >> 16) & 1u)) >> 16;   // round-to-nearest-even
  return (unsigned short)r;
}
__device__ inline unsigned pack2(float a, float b) {
  return (unsigned)f2bf(a) | ((unsigned)f2bf(b) << 16);
}
__device__ inline float wred_sum(float v) {
#pragma unroll
  for (int m = 16; m; m >>= 1) v += __shfl_xor(v, m, 32);
  return v;
}
__device__ inline float wred_max(float v) {
#pragma unroll
  for (int m = 16; m; m >>= 1) v = fmaxf(v, __shfl_xor(v, m, 32));
  return v;
}

// A/B bf16 fragment (16x32 / 32x16) from an LDS panel with 32-short rows.
// Layout per ISA 7.12.2: lane<16 -> row m0+lane, K 0..7 & 16..23;
//                        lane>=16 -> row m0+lane-16, K 8..15 & 24..31.
__device__ inline v16bf ldfrag32(const unsigned short (*p)[32], int row0) {
  int lane = threadIdx.x & 31;
  int row  = row0 + (lane & 15);
  int c0   = (lane < 16) ? 0 : 8;
  FragU f;
  f.q[0] = *(const uint4*)&p[row][c0];
  f.q[1] = *(const uint4*)&p[row][c0 + 16];
  return f.v;
}

__device__ inline v8f wmma_bf16(v16bf a, v16bf b, v8f c) {
  return __builtin_amdgcn_wmma_f32_16x16x32_bf16(false, a, false, b, (short)0, c,
                                                 false, false);
}

// ---------------- k0: per-(b,n) row reductions + x_norm -------------------
__global__ __launch_bounds__(256)
void k0_prep(const float* __restrict__ x_in, const float* __restrict__ pp,
             float* xener, float* absx, float* xnorm,
             float* ppsum, float* ppsq) {
  int bn = blockIdx.x;                  // b*32+n
  int lane = threadIdx.x & 31;
  int wave = threadIdx.x >> 5;          // 0..7
  __shared__ float red[256];
  const float* xb = x_in + (size_t)bn * F_ * HID_;
  const float* pb = pp   + (size_t)bn * F_ * HID_;
  for (int f = wave; f < F_; f += 8) {
    float x = xb[f * HID_ + lane];
    float p = pb[f * HID_ + lane];
    float sx  = wred_sum(x);
    float sax = wred_sum(fabsf(x));
    float sp  = wred_sum(p);
    float sp2 = wred_sum(p * p);
    if (lane == 0) {
      xener[bn * F_ + f] = sx * (1.0f / HID_);
      absx [bn * F_ + f] = sax;
      ppsum[bn * F_ + f] = sp;
      ppsq [bn * F_ + f] = sp2;
    }
  }
  __threadfence_block();
  __syncthreads();
  float m = 0.f;
  for (int f = threadIdx.x; f < F_; f += 256) m = fmaxf(m, fabsf(xener[bn * F_ + f]));
  red[threadIdx.x] = m;
  __syncthreads();
  for (int s = 128; s > 0; s >>= 1) {
    if (threadIdx.x < s) red[threadIdx.x] = fmaxf(red[threadIdx.x], red[threadIdx.x + s]);
    __syncthreads();
  }
  float denom = red[0] + 1e-8f;
  for (int f = threadIdx.x; f < F_; f += 256) {
    float v = xener[bn * F_ + f] / denom;
    xnorm[bn * F_ + f] = fminf(fmaxf(v, -0.99f), 0.99f);
  }
}

// ---------------- k1: coeff + edge energies -> mask, gate -----------------
__global__ __launch_bounds__(256)
void k1_coeff(const float* __restrict__ xnorm, const float* __restrict__ absx,
              const float* __restrict__ spw,  const float* __restrict__ bscale,
              const float* __restrict__ tau,  const float* __restrict__ temp,
              const float* __restrict__ omiga, const float* __restrict__ grid,
              float* coeff, float* mask, float* sgate) {
  int bn = blockIdx.x; int n = bn & 31;
  int lane = threadIdx.x & 31;          // output neuron o
  int wave = threadIdx.x >> 5;
  __shared__ float red[8][32];
  const float* sw = spw + (n * 32 + lane) * NC_;
  float bs = bscale[n * 32 + lane];
  float om = fabsf(omiga[n * 32 + lane]);
  float acc = 0.f;
  for (int f = wave; f < F_; f += 8) {
    float xr  = xnorm[bn * F_ + f];
    float sil = xr / (1.f + __expf(-xr));
    float s   = sil * bs;
    for (int c = 0; c < NC_; ++c) {          // cubic B-spline basis, h = 2/34
      float d = fabsf(xr - grid[c]) * 17.0f;
      float bsis;
      if (d < 1.f)      bsis = (2.f / 3.f) - d * d + d * d * d * 0.5f;
      else if (d < 2.f) { float t2 = 2.f - d; bsis = t2 * t2 * t2 * (1.f / 6.f); }
      else              bsis = 0.f;
      s += bsis * sw[c];
    }
    float co = s + om;
    coeff[((size_t)bn * 32 + lane) * F_ + f] = co;
    acc += fabsf(co) * absx[bn * F_ + f];
  }
  red[wave][lane] = acc;
  __syncthreads();
  if (wave == 0) {
    float ee = 0.f;
    for (int w2 = 0; w2 < 8; ++w2) ee += red[w2][lane];
    ee *= (1.0f / (F_ * HID_));
    float ta = fabsf(tau[n * 32 + lane]);
    float tv = fabsf(temp[0]) + 1e-4f;
    float mk = 1.f / (1.f + __expf(-(ee - ta) / tv));
    mask [bn * 32 + lane] = mk;
    sgate[bn * 32 + lane] = (ee / (ta + 1e-8f)) * mk;
  }
}

// ---------------- k2: analytic LayerNorm stats for K and Q rows -----------
__global__ __launch_bounds__(256)
void k2_lnstats(const float* __restrict__ sgate, const float* __restrict__ ppsum,
                const float* __restrict__ ppsq,
                float* meanK, float* rstdK, float* meanQ, float* rstdQ) {
  int bo = blockIdx.x; int b = bo >> 5, o = bo & 31;
  int f = threadIdx.x;
  if (f >= F_) return;
  float mk = 0, k2 = 0, mq = 0, q2 = 0;
  for (int i = 0; i < 16; ++i) {
    float sK = sgate[(b * 32 + i) * 32 + o];
    float sQ = sgate[(b * 32 + i + 16) * 32 + o];
    mk += sK * ppsum[(b * 32 + i) * F_ + f];
    k2 += sK * sK * ppsq[(b * 32 + i) * F_ + f];
    mq += sQ * ppsum[(b * 32 + i + 16) * F_ + f];
    q2 += sQ * sQ * ppsq[(b * 32 + i + 16) * F_ + f];
  }
  float mK = mk * (1.f / 512.f), mQ = mq * (1.f / 512.f);
  float vK = fmaxf(k2 * (1.f / 512.f) - mK * mK, 0.f);
  float vQ = fmaxf(q2 * (1.f / 512.f) - mQ * mQ, 0.f);
  meanK[bo * F_ + f] = mK; rstdK[bo * F_ + f] = rsqrtf(vK + 1e-5f);
  meanQ[bo * F_ + f] = mQ; rstdQ[bo * F_ + f] = rsqrtf(vQ + 1e-5f);
}

// ------- k3: combined (mean over n) + LN + WMMA proj GEMM (xn @ W2[o]) ----
__global__ __launch_bounds__(256)
void k3_combined_proj(const float* __restrict__ x_in, const float* __restrict__ coeff,
                      const float* __restrict__ mask, const float* __restrict__ W2,
                      const float* __restrict__ ln_w, const float* __restrict__ ln_b,
                      float* combined, float* proj_out) {
  int bo = blockIdx.x; int b = bo >> 5, o = bo & 31;
  int lane = threadIdx.x & 31;           // d
  int wave = threadIdx.x >> 5;           // 0..7
  __shared__ __attribute__((aligned(16))) unsigned short xnP[FP_][32];
  __shared__ __attribute__((aligned(16))) unsigned short w2T[32][32];
  // stage W2^T as bf16: w2T[h][d] = W2[o][d][h]
  for (int idx = threadIdx.x; idx < 1024; idx += 256) {
    int h = idx >> 5, d = idx & 31;
    w2T[h][d] = f2bf(W2[((size_t)o * 32 + d) * 32 + h]);
  }
  // zero the padded rows 187..191
  for (int idx = threadIdx.x; idx < (FP_ - F_) * 32; idx += 256)
    xnP[F_ + idx / 32][idx & 31] = 0;
  float lw = ln_w[o * 32 + lane], lb = ln_b[o * 32 + lane];
  for (int f = wave; f < F_; f += 8) {
    float acc = 0.f;
    for (int n = 0; n < 32; ++n) {
      float cm = coeff[((size_t)(b * 32 + n) * 32 + o) * F_ + f] *
                 mask[(b * 32 + n) * 32 + o];
      acc += cm * x_in[((size_t)(b * 32 + n) * F_ + f) * 32 + lane];
    }
    float cb = acc * (1.f / 32.f);
    combined[((size_t)bo * F_ + f) * 32 + lane] = cb;
    float mu  = wred_sum(cb) * (1.f / 32.f);
    float dv  = cb - mu;
    float var = wred_sum(dv * dv) * (1.f / 32.f);
    xnP[f][lane] = f2bf(dv * rsqrtf(var + 1e-5f) * lw + lb);
  }
  __syncthreads();
  // proj[192x32] = xn[192x32] @ W2[32x32], one K=32 WMMA step per tile
  for (int mt = wave; mt < 12; mt += 8) {
    v16bf a  = ldfrag32(xnP, mt * 16);
    v16bf b0 = ldfrag32(w2T, 0);
    v16bf b1 = ldfrag32(w2T, 16);
    v8f c0 = {}; v8f c1 = {};
    c0 = wmma_bf16(a, b0, c0);
    c1 = wmma_bf16(a, b1, c1);
    int rbase = mt * 16 + ((lane >> 4) << 3);
    int col   = lane & 15;
#pragma unroll
    for (int r = 0; r < 8; ++r) {
      int row = rbase + r;
      if (row < F_) {
        proj_out[((size_t)bo * F_ + row) * 32 + col]      = c0[r];
        proj_out[((size_t)bo * F_ + row) * 32 + col + 16] = c1[r];
      }
    }
  }
}

// ------- k4: attention scores S = Kn(187x512) @ Qn^T via bf16 WMMA --------
// 16 waves; wave (mg,ng) owns a 3x3 block of 16x16 C tiles (144 total).
// K/Q panels (32 k-columns each) normalized on the fly into LDS per k-step.
__global__ __launch_bounds__(512)
void k4_attn(const float* __restrict__ pp, const float* __restrict__ sgate,
             const float* __restrict__ meanK, const float* __restrict__ rstdK,
             const float* __restrict__ meanQ, const float* __restrict__ rstdQ,
             float* sbuf) {
  int bo = blockIdx.x; int b = bo >> 5, o = bo & 31;
  int lane = threadIdx.x & 31;
  int wave = threadIdx.x >> 5;          // 0..15
  int mg = wave >> 2, ng = wave & 3;
  __shared__ __attribute__((aligned(16))) unsigned short Kp[FP_][32];
  __shared__ __attribute__((aligned(16))) unsigned short Qp[FP_][32];
  v8f acc[3][3];
#pragma unroll
  for (int j = 0; j < 3; ++j)
#pragma unroll
    for (int jj = 0; jj < 3; ++jj) acc[j][jj] = (v8f){};

  int h  = lane;
  int fr = threadIdx.x >> 5;            // row group 0..15 for panel fill
  for (int i = 0; i < 16; ++i) {        // k-panels: k = i*32 + h
    float sK = sgate[(b * 32 + i) * 32 + o];
    float sQ = sgate[(b * 32 + i + 16) * 32 + o];
    const float* pK = pp + (size_t)(b * 32 + i) * F_ * 32;
    const float* pQ = pp + (size_t)(b * 32 + i + 16) * F_ * 32;
    for (int f = fr; f < FP_; f += 16) {
      unsigned short kv = 0, qv = 0;
      if (f < F_) {
        kv = f2bf((sK * pK[f * 32 + h] - meanK[bo * F_ + f]) * rstdK[bo * F_ + f]);
        qv = f2bf((sQ * pQ[f * 32 + h] - meanQ[bo * F_ + f]) * rstdQ[bo * F_ + f]);
      }
      Kp[f][h] = kv;
      Qp[f][h] = qv;
    }
    __syncthreads();
    v16bf a0 = ldfrag32(Kp, (mg * 3 + 0) * 16);
    v16bf a1 = ldfrag32(Kp, (mg * 3 + 1) * 16);
    v16bf a2 = ldfrag32(Kp, (mg * 3 + 2) * 16);
    v16bf b0 = ldfrag32(Qp, (ng * 3 + 0) * 16);
    v16bf b1 = ldfrag32(Qp, (ng * 3 + 1) * 16);
    v16bf b2 = ldfrag32(Qp, (ng * 3 + 2) * 16);
    acc[0][0] = wmma_bf16(a0, b0, acc[0][0]);
    acc[0][1] = wmma_bf16(a0, b1, acc[0][1]);
    acc[0][2] = wmma_bf16(a0, b2, acc[0][2]);
    acc[1][0] = wmma_bf16(a1, b0, acc[1][0]);
    acc[1][1] = wmma_bf16(a1, b1, acc[1][1]);
    acc[1][2] = wmma_bf16(a1, b2, acc[1][2]);
    acc[2][0] = wmma_bf16(a2, b0, acc[2][0]);
    acc[2][1] = wmma_bf16(a2, b1, acc[2][1]);
    acc[2][2] = wmma_bf16(a2, b2, acc[2][2]);
    __syncthreads();
  }
  float* sb = sbuf + (size_t)bo * FP_ * FP_;
#pragma unroll
  for (int j = 0; j < 3; ++j)
#pragma unroll
    for (int jj = 0; jj < 3; ++jj) {
      int m0 = (mg * 3 + j) * 16, n0 = (ng * 3 + jj) * 16;
      int rbase = m0 + ((lane >> 4) << 3);
      int col   = n0 + (lane & 15);
#pragma unroll
      for (int r = 0; r < 8; ++r)
        sb[(size_t)(rbase + r) * FP_ + col] = acc[j][jj][r];
    }
}

// ---------------- k5: in-place row softmax with 1/(sqrt(512)*t) -----------
__global__ __launch_bounds__(256)
void k5_softmax(float* sbuf, const float* __restrict__ temp) {
  int bo = blockIdx.x;
  int lane = threadIdx.x & 31, wave = threadIdx.x >> 5;
  float tv = fabsf(temp[0]) + 1e-4f;
  float scale = 1.f / (sqrtf(512.f) * tv);
  float* sb = sbuf + (size_t)bo * FP_ * FP_;
  for (int f = wave; f < F_; f += 8) {
    float* row = sb + (size_t)f * FP_;
    float m = -3.4e38f;
    for (int g = lane; g < F_; g += 32) m = fmaxf(m, row[g]);
    m = wred_max(m);
    float s = 0.f;
    for (int g = lane; g < F_; g += 32) s += __expf((row[g] - m) * scale);
    s = wred_sum(s);
    float inv = 1.f / s;
    for (int g = lane; g < F_; g += 32) row[g] = __expf((row[g] - m) * scale) * inv;
  }
}

// ------- k6: out = |beta|*(W1@BM) + |alpha|*BM + gamma*combined (WMMA) ----
__global__ __launch_bounds__(384)
void k6_out(const float* __restrict__ sbuf, const float* __restrict__ proj_out,
            const float* __restrict__ b_param, const float* __restrict__ combined,
            const float* __restrict__ beta, const float* __restrict__ alpha_,
            const float* __restrict__ gamma_, float* final_out) {
  int bo = blockIdx.x; int o = bo & 31;
  int lane = threadIdx.x & 31, wave = threadIdx.x >> 5;  // 12 waves, wave == m-tile
  __shared__ __attribute__((aligned(16))) unsigned short BMT[32][FP_]; // BM^T bf16
  for (int idx = threadIdx.x; idx < 32 * FP_; idx += 384) {
    int hh = idx / FP_, g = idx % FP_;
    float v = 0.f;
    if (g < F_)
      v = proj_out[((size_t)bo * F_ + g) * 32 + hh] +
          b_param[((size_t)o * F_ + g) * 32 + hh];
    BMT[hh][g] = f2bf(v);
  }
  __syncthreads();
  const float* sb = sbuf + (size_t)bo * FP_ * FP_;
  int mt = wave;
  v8f c0 = {}; v8f c1 = {};
  for (int kt = 0; kt < 6; ++kt) {      // K = 192 in steps of 32
    // A fragment: fp32 probabilities from sbuf -> bf16
    int row = mt * 16 + (lane & 15);
    const float* rp = sb + (size_t)row * FP_;
    int cb = kt * 32 + ((lane < 16) ? 0 : 8);
    FragU a;
    a.q[0] = make_uint4(pack2(rp[cb + 0],  rp[cb + 1]),
                        pack2(rp[cb + 2],  rp[cb + 3]),
                        pack2(rp[cb + 4],  rp[cb + 5]),
                        pack2(rp[cb + 6],  rp[cb + 7]));
    a.q[1] = make_uint4(pack2(rp[cb + 16], rp[cb + 17]),
                        pack2(rp[cb + 18], rp[cb + 19]),
                        pack2(rp[cb + 20], rp[cb + 21]),
                        pack2(rp[cb + 22], rp[cb + 23]));
    // B fragments from BMT (rows = output column h)
    int nr = lane & 15;
    FragU bb0, bb1;
    bb0.q[0] = *(const uint4*)&BMT[nr][cb];
    bb0.q[1] = *(const uint4*)&BMT[nr][cb + 16];
    bb1.q[0] = *(const uint4*)&BMT[nr + 16][cb];
    bb1.q[1] = *(const uint4*)&BMT[nr + 16][cb + 16];
    c0 = wmma_bf16(a.v, bb0.v, c0);
    c1 = wmma_bf16(a.v, bb1.v, c1);
  }
  float bA = fabsf(beta[o]), aA = fabsf(alpha_[o]), gm = gamma_[o];
  int rbase = mt * 16 + ((lane >> 4) << 3);
  int col   = lane & 15;
#pragma unroll
  for (int r = 0; r < 8; ++r) {
    int row = rbase + r;
    if (row >= F_) continue;
    size_t base = ((size_t)bo * F_ + row) * 32;
    size_t bmb  = ((size_t)o  * F_ + row) * 32;
    float bm0 = proj_out[base + col]      + b_param[bmb + col];
    float bm1 = proj_out[base + col + 16] + b_param[bmb + col + 16];
    final_out[base + col]      = bA * c0[r] + aA * bm0 + gm * combined[base + col];
    final_out[base + col + 16] = bA * c1[r] + aA * bm1 + gm * combined[base + col + 16];
  }
}

// --------------------------------------------------------------------------
extern "C" void kernel_launch(void* const* d_in, const int* in_sizes, int n_in,
                              void* d_out, int out_size, void* d_ws, size_t ws_size,
                              hipStream_t stream) {
  (void)in_sizes; (void)n_in; (void)out_size; (void)ws_size;
  const float* x_in = (const float*)d_in[0];
  const float* pp   = (const float*)d_in[1];
  const float* spw  = (const float*)d_in[2];
  const float* bsc  = (const float*)d_in[3];
  const float* tau  = (const float*)d_in[4];
  const float* temp = (const float*)d_in[5];
  const float* omg  = (const float*)d_in[6];
  const float* grid = (const float*)d_in[7];
  const float* W2   = (const float*)d_in[8];
  const float* bprm = (const float*)d_in[9];
  const float* lnw  = (const float*)d_in[10];
  const float* lnb  = (const float*)d_in[11];
  const float* beta = (const float*)d_in[12];
  const float* gam  = (const float*)d_in[13];
  const float* alph = (const float*)d_in[14];

  float* out       = (float*)d_out;
  float* final_out = out;                                         // [B,O,F,H]
  float* proj_out  = out + (size_t)B_ * NOUT_ * F_ * HID_;        // [B,O,F,H]

  // Workspace carve-out (~52 MB total)
  float* w = (float*)d_ws;
  size_t off = 0;
  auto alloc = [&](size_t n) { float* p = w + off; off += n; return p; };
  const size_t BNF = (size_t)B_ * NIN_ * F_;
  const size_t BOF = (size_t)B_ * NOUT_ * F_;
  float* xener = alloc(BNF);
  float* absx  = alloc(BNF);
  float* xnorm = alloc(BNF);
  float* ppsum = alloc(BNF);
  float* ppsq  = alloc(BNF);
  float* coeff = alloc((size_t)B_ * NIN_ * NOUT_ * F_);
  float* maskb = alloc((size_t)B_ * NIN_ * NOUT_);
  float* sgate = alloc((size_t)B_ * NIN_ * NOUT_);
  float* meanK = alloc(BOF);
  float* rstdK = alloc(BOF);
  float* meanQ = alloc(BOF);
  float* rstdQ = alloc(BOF);
  float* comb  = alloc((size_t)B_ * NOUT_ * F_ * HID_);
  float* sbuf  = alloc((size_t)B_ * NOUT_ * FP_ * FP_);

  k0_prep<<<B_ * NIN_, 256, 0, stream>>>(x_in, pp, xener, absx, xnorm, ppsum, ppsq);
  k1_coeff<<<B_ * NIN_, 256, 0, stream>>>(xnorm, absx, spw, bsc, tau, temp, omg, grid,
                                          coeff, maskb, sgate);
  k2_lnstats<<<B_ * NOUT_, 256, 0, stream>>>(sgate, ppsum, ppsq,
                                             meanK, rstdK, meanQ, rstdQ);
  k3_combined_proj<<<B_ * NOUT_, 256, 0, stream>>>(x_in, coeff, maskb, W2, lnw, lnb,
                                                   comb, proj_out);
  k4_attn<<<B_ * NOUT_, 512, 0, stream>>>(pp, sgate, meanK, rstdK, meanQ, rstdQ, sbuf);
  k5_softmax<<<B_ * NOUT_, 256, 0, stream>>>(sbuf, temp);
  k6_out<<<B_ * NOUT_, 384, 0, stream>>>(sbuf, proj_out, bprm, comb,
                                         beta, alph, gam, final_out);
}
